// CrossModalAttention_42039139893703
// MI455X (gfx1250) — matrix operand order repaired and checked by reading, not compiled, not run on hardware
//
#include <hip/hip_runtime.h>

// ---- sizes (compile-time for this problem) ----
#define HEADS 8
#define DHEAD 64
#define DIM 512
#define BATCH 4
#define NTOK 2048
#define TOKS (BATCH * NTOK)      // 8192
#define SCALE 0.125f             // 1/sqrt(64)

typedef __attribute__((ext_vector_type(16))) __bf16 v16bf;
typedef __attribute__((ext_vector_type(8)))  __bf16 v8bf;
typedef __attribute__((ext_vector_type(8)))  float  v8f;
typedef __attribute__((ext_vector_type(4)))  float  v4f;
typedef int gv4i __attribute__((vector_size(16)));   // matches async-LDS builtin param

#if defined(__HIP_DEVICE_COMPILE__) && \
    __has_builtin(__builtin_amdgcn_global_load_async_to_lds_b128) && \
    __has_builtin(__builtin_amdgcn_s_wait_asynccnt)
#define USE_ASYNC_LDS 1
#else
#define USE_ASYNC_LDS 0
#endif

static __device__ __forceinline__ v8f wmma_bf16(v16bf a, v16bf b, v8f c) {
  // D = A(16x32 bf16) x B(32x16 bf16) + C(16x16 f32)
  return __builtin_amdgcn_wmma_f32_16x16x32_bf16(false, a, false, b, (short)0, c,
                                                 false, false);
}

static __device__ __forceinline__ v8f zero8() {
  v8f z;
#pragma unroll
  for (int i = 0; i < 8; ++i) z[i] = 0.0f;
  return z;
}

static __device__ __forceinline__ v16bf cat8(v8bf lo, v8bf hi) {
  v16bf r;
#pragma unroll
  for (int i = 0; i < 8; ++i) { r[i] = lo[i]; r[i + 8] = hi[i]; }
  return r;
}

// A-operand chunk (16x32 bf16): lane (li,g) holds row M=li,
// K=kk+16g..kk+16g+15 contiguous (matches B operand order)
static __device__ __forceinline__ v16bf ldA_bf16(const __bf16* row, int kk, int g) {
  v8bf lo = *(const v8bf*)(row + kk + 16 * g);
  v8bf hi = *(const v8bf*)(row + kk + 16 * g + 8);
  return cat8(lo, hi);
}

// B-operand chunk (32x16 bf16): lane (li,g) holds column N=li, K=16g..16g+15 contiguous
static __device__ __forceinline__ v16bf ldB_bf16(const __bf16* p) {
  v8bf lo = *(const v8bf*)p;
  v8bf hi = *(const v8bf*)(p + 8);
  return cat8(lo, hi);
}

// ------------------------------------------------------------------
// Kernel 0: one-time f32 -> bf16 conversion (x and the 4 weight matrices).
// ------------------------------------------------------------------
__global__ void __launch_bounds__(256)
xattn_cvt_kernel(const float* __restrict__ src, __bf16* __restrict__ dst, int n) {
  const int i = (blockIdx.x * 256 + threadIdx.x) * 8;
  if (i + 8 <= n) {
    v4f f0 = *(const v4f*)(src + i);
    v4f f1 = *(const v4f*)(src + i + 4);
    v8bf r;
#pragma unroll
    for (int k = 0; k < 4; ++k) { r[k] = (__bf16)f0[k]; r[k + 4] = (__bf16)f1[k]; }
    *(v8bf*)(dst + i) = r;
  }
}

// ------------------------------------------------------------------
// Kernel 1: QKV projection (all-bf16 operands), K-loop software-pipelined
// by one stage so operand loads overlap the previous chunk's WMMAs.
// q[h,b,n,dh], k[h,b,n,dh], vT[h,b,dh,n] (bf16)
// ------------------------------------------------------------------
__global__ void __launch_bounds__(256)
xattn_qkv_kernel(const __bf16* __restrict__ xb,
                 const __bf16* __restrict__ wqb, const __bf16* __restrict__ wkb,
                 const __bf16* __restrict__ wvb,
                 __bf16* __restrict__ qb, __bf16* __restrict__ kb,
                 __bf16* __restrict__ vtb) {
  const int lane = threadIdx.x & 31;
  const int li = lane & 15, g = lane >> 4;
  const int task = blockIdx.x * 8 + (threadIdx.x >> 5);  // 12288 tasks
  const int dg = task % 24;          // 3 weights x 8 heads
  const int tti = task / 24;         // 512 t-tiles
  const int w = dg >> 3, h = dg & 7;
  const __bf16* W = (w == 0) ? wqb : ((w == 1) ? wkb : wvb);
  const int t0 = tti * 16;
  const int b = t0 >> 11, n0 = t0 & (NTOK - 1);
  const int d0 = h * DHEAD;

  v8f acc[4] = {zero8(), zero8(), zero8(), zero8()};
  const __bf16* arow = xb + (size_t)(t0 + li) * DIM;
  const __bf16* wrow[4];
#pragma unroll
  for (int j = 0; j < 4; ++j)
    wrow[j] = W + (size_t)(d0 + j * 16 + li) * DIM + 16 * g;

  // prologue: load chunk kk=0
  v16bf a = ldA_bf16(arow, 0, g);
  v16bf bt[4];
#pragma unroll
  for (int j = 0; j < 4; ++j) bt[j] = ldB_bf16(wrow[j]);

#pragma unroll 2
  for (int kk = 0; kk < DIM; kk += 32) {
    const int kn = kk + 32;
    v16bf an = a, bn[4] = {bt[0], bt[1], bt[2], bt[3]};
    if (kn < DIM) {  // issue next chunk's loads before this chunk's WMMAs
      an = ldA_bf16(arow, kn, g);
#pragma unroll
      for (int j = 0; j < 4; ++j) bn[j] = ldB_bf16(wrow[j] + kn);
    }
#pragma unroll
    for (int j = 0; j < 4; ++j) acc[j] = wmma_bf16(a, bt[j], acc[j]);
    a = an;
#pragma unroll
    for (int j = 0; j < 4; ++j) bt[j] = bn[j];
  }

  // C layout: VGPR r, lane (li,g) -> row r+8g, col li
  if (w < 2) {
    __bf16* base = ((w == 0) ? qb : kb) + ((size_t)(h * BATCH + b) * NTOK) * DHEAD;
#pragma unroll
    for (int j = 0; j < 4; ++j)
#pragma unroll
      for (int r = 0; r < 8; ++r)
        base[(size_t)(n0 + r + 8 * g) * DHEAD + j * 16 + li] = (__bf16)acc[j][r];
  } else {
    __bf16* base = vtb + ((size_t)(h * BATCH + b) * DHEAD) * NTOK;
#pragma unroll
    for (int j = 0; j < 4; ++j)
#pragma unroll
      for (int r = 0; r < 8; ++r)
        base[(size_t)(j * 16 + li) * NTOK + n0 + r + 8 * g] = (__bf16)acc[j][r];
  }
}

// ------------------------------------------------------------------
// Kernel 2: fused flash attention. One wave per (h,b,16-query tile); all 8
// waves of a block share (h,b) and cooperatively double-buffer the 32-key
// K/V tiles into LDS via async-to-LDS loads (ASYNCcnt-tracked).
// mask fill = constant 1e-9 BEFORE softmax (faithful to reference).
// ------------------------------------------------------------------
__global__ void __launch_bounds__(256)
xattn_flash_kernel(const __bf16* __restrict__ qb, const __bf16* __restrict__ kb,
                   const __bf16* __restrict__ vtb, const int* __restrict__ mask,
                   __bf16* __restrict__ ob) {
  const int tid = threadIdx.x;
  const int lane = tid & 31;
  const int li = lane & 15, g = lane >> 4;
  const int wave = tid >> 5;
  const int hb = blockIdx.x >> 4;                 // 32 (h,b) pairs, 16 blocks each
  const int h = hb >> 2, b = hb & 3;
  const int tt = (blockIdx.x & 15) * 8 + wave;    // 128 query tiles per (h,b)
  const int t0 = tt * 16;

  // K stage: 32 keys x 64 dh  (row stride 72 elems = 144B, 16B aligned)
  // V stage: 64 dh  x 32 keys (row stride 40 elems =  80B, 16B aligned)
  __shared__ __align__(16) __bf16 ksh[2][32 * 72];
  __shared__ __align__(16) __bf16 vsh[2][64 * 40];
  __shared__ __align__(16) __bf16 psh[8 * 16 * 40];  // per-wave P transpose tile
  __bf16* pst = &psh[wave * 16 * 40];

  // cooperative staging geometry: 256 threads x 16B cover each 4KB tile
  const int krow = tid >> 3, kch = (tid & 7) * 8;  // 32 rows x 8 chunks
  const int vrow = tid >> 2, vch = (tid & 3) * 8;  // 64 rows x 4 chunks

  const __bf16* kbase = kb + ((size_t)(h * BATCH + b) * NTOK) * DHEAD;
  const __bf16* vbase = vtb + ((size_t)(h * BATCH + b) * DHEAD) * NTOK;
  const int* mrow = mask + b * NTOK;

  // Q A-operands for K(dh)=0..31 and 32..63, resident for the whole loop
  const __bf16* qrow = qb + ((size_t)(h * BATCH + b) * NTOK + t0 + li) * DHEAD;
  const v16bf qa0 = ldA_bf16(qrow, 0, g);
  const v16bf qa1 = ldA_bf16(qrow, 32, g);

  v8f o[4] = {zero8(), zero8(), zero8(), zero8()};
  float m[8], l[8];
#pragma unroll
  for (int r = 0; r < 8; ++r) { m[r] = -1e30f; l[r] = 0.0f; }

  // ---- stage one 32-key K/V tile into LDS buffer `buf` ----
  auto stage = [&](int buf, int s0) {
    const __bf16* kg = kbase + (size_t)(s0 + krow) * DHEAD + kch;
    const __bf16* vg = vbase + (size_t)vrow * NTOK + s0 + vch;
    __bf16* kl = &ksh[buf][krow * 72 + kch];
    __bf16* vl = &vsh[buf][vrow * 40 + vch];
#if USE_ASYNC_LDS
    gv4i* kg4 = (gv4i*)kg;
    gv4i* vg4 = (gv4i*)vg;
    gv4i* kl4 = (gv4i*)kl;
    gv4i* vl4 = (gv4i*)vl;
    __builtin_amdgcn_global_load_async_to_lds_b128(
        (__attribute__((address_space(1))) gv4i*)kg4,
        (__attribute__((address_space(3))) gv4i*)kl4, 0, 0);
    __builtin_amdgcn_global_load_async_to_lds_b128(
        (__attribute__((address_space(1))) gv4i*)vg4,
        (__attribute__((address_space(3))) gv4i*)vl4, 0, 0);
#else
    *(v8bf*)kl = *(const v8bf*)kg;
    *(v8bf*)vl = *(const v8bf*)vg;
#endif
  };

  stage(0, 0);

  for (int s0 = 0; s0 < NTOK; s0 += 32) {
    const int cur = (s0 >> 5) & 1;
    __syncthreads();  // all reads of buf cur^1 finished before it is overwritten
    if (s0 + 32 < NTOK) {
      stage(cur ^ 1, s0 + 32);
#if USE_ASYNC_LDS
      __builtin_amdgcn_s_wait_asynccnt(2);  // own current-tile ops complete
#endif
    } else {
#if USE_ASYNC_LDS
      __builtin_amdgcn_s_wait_asynccnt(0);
#endif
    }
    __syncthreads();  // everyone's current-tile data visible in LDS

    const int mk0 = mrow[s0 + li];
    const int mk1 = mrow[s0 + 16 + li];

    // S = Q Kt : two 16x16 tiles (st=0: cols s0..s0+15, st=1: +16..31)
    v8f sa[2];
#pragma unroll
    for (int st = 0; st < 2; ++st) {
      const __bf16* krw = &ksh[cur][(st * 16 + li) * 72];
      v8f s = zero8();
      s = wmma_bf16(qa0, ldB_bf16(krw + 16 * g), s);       // dh 0..31
      s = wmma_bf16(qa1, ldB_bf16(krw + 32 + 16 * g), s);  // dh 32..63
      sa[st] = s;
    }

    // scale + constant mask fill, online softmax update
    float sv[2][8], alpha[8];
#pragma unroll
    for (int r = 0; r < 8; ++r) {
      sv[0][r] = mk0 ? sa[0][r] * SCALE : 1e-9f;
      sv[1][r] = mk1 ? sa[1][r] * SCALE : 1e-9f;
      float mx = fmaxf(sv[0][r], sv[1][r]);
      mx = fmaxf(mx, __shfl_xor(mx, 1, 16));
      mx = fmaxf(mx, __shfl_xor(mx, 2, 16));
      mx = fmaxf(mx, __shfl_xor(mx, 4, 16));
      mx = fmaxf(mx, __shfl_xor(mx, 8, 16));
      const float mn = fmaxf(m[r], mx);
      alpha[r] = __expf(m[r] - mn);
      m[r] = mn;
      const float p0 = __expf(sv[0][r] - mn);
      const float p1 = __expf(sv[1][r] - mn);
      sv[0][r] = p0; sv[1][r] = p1;
      float rs = p0 + p1;
      rs += __shfl_xor(rs, 1, 16);
      rs += __shfl_xor(rs, 2, 16);
      rs += __shfl_xor(rs, 4, 16);
      rs += __shfl_xor(rs, 8, 16);
      l[r] = l[r] * alpha[r] + rs;
    }
#pragma unroll
    for (int j = 0; j < 4; ++j)
#pragma unroll
      for (int r = 0; r < 8; ++r) o[j][r] *= alpha[r];

    // P: C-layout -> per-wave LDS tile (row-major 16x32 bf16) -> A-layout
#pragma unroll
    for (int st = 0; st < 2; ++st)
#pragma unroll
      for (int r = 0; r < 8; ++r)
        pst[(r + 8 * g) * 40 + st * 16 + li] = (__bf16)sv[st][r];
    asm volatile("" ::: "memory");  // keep ds_load after ds_store; HW keeps DS in order
    const v16bf pa = cat8(*(const v8bf*)(pst + li * 40 + 16 * g),
                          *(const v8bf*)(pst + li * 40 + 16 * g + 8));

    // O += P V : contraction over s (32), four 16-wide dh tiles
#pragma unroll
    for (int j = 0; j < 4; ++j) {
      const __bf16* vrw = &vsh[cur][(j * 16 + li) * 40 + 16 * g];
      o[j] = wmma_bf16(pa, ldB_bf16(vrw), o[j]);
    }
  }

  // normalize and store head-major channels O[b,n, h*64+dh]
  float inv[8];
#pragma unroll
  for (int r = 0; r < 8; ++r) inv[r] = 1.0f / l[r];
#pragma unroll
  for (int j = 0; j < 4; ++j)
#pragma unroll
    for (int r = 0; r < 8; ++r)
      ob[(size_t)(b * NTOK + t0 + r + 8 * g) * DIM + h * DHEAD + j * 16 + li] =
          (__bf16)(o[j][r] * inv[r]);
}

// ------------------------------------------------------------------
// Kernel 3: output projection  out[t,e] = sum_d O[t,d] * Wout[e,d]  (f32 out),
// same one-stage software pipeline as kernel 1.
// ------------------------------------------------------------------
__global__ void __launch_bounds__(256)
xattn_proj_kernel(const __bf16* __restrict__ ob, const __bf16* __restrict__ wob,
                  float* __restrict__ out) {
  const int lane = threadIdx.x & 31;
  const int li = lane & 15, g = lane >> 4;
  const int task = blockIdx.x * 8 + (threadIdx.x >> 5);  // 4096 tasks
  const int eg = task & 7;
  const int tti = task >> 3;
  const int t0 = tti * 16, e0 = eg * 64;

  v8f acc[4] = {zero8(), zero8(), zero8(), zero8()};
  const __bf16* arow = ob + (size_t)(t0 + li) * DIM;
  const __bf16* wrow[4];
#pragma unroll
  for (int j = 0; j < 4; ++j)
    wrow[j] = wob + (size_t)(e0 + j * 16 + li) * DIM + 16 * g;

  v16bf a = ldA_bf16(arow, 0, g);
  v16bf bt[4];
#pragma unroll
  for (int j = 0; j < 4; ++j) bt[j] = ldB_bf16(wrow[j]);

#pragma unroll 2
  for (int kk = 0; kk < DIM; kk += 32) {
    const int kn = kk + 32;
    v16bf an = a, bn[4] = {bt[0], bt[1], bt[2], bt[3]};
    if (kn < DIM) {
      an = ldA_bf16(arow, kn, g);
#pragma unroll
      for (int j = 0; j < 4; ++j) bn[j] = ldB_bf16(wrow[j] + kn);
    }
#pragma unroll
    for (int j = 0; j < 4; ++j) acc[j] = wmma_bf16(a, bt[j], acc[j]);
    a = an;
#pragma unroll
    for (int j = 0; j < 4; ++j) bt[j] = bn[j];
  }

#pragma unroll
  for (int j = 0; j < 4; ++j)
#pragma unroll
    for (int r = 0; r < 8; ++r)
      out[(size_t)(t0 + r + 8 * g) * DIM + e0 + j * 16 + li] = acc[j][r];
}

// ------------------------------------------------------------------
extern "C" void kernel_launch(void* const* d_in, const int* in_sizes, int n_in,
                              void* d_out, int out_size, void* d_ws, size_t ws_size,
                              hipStream_t stream) {
  const float* x    = (const float*)d_in[0];
  const int*   mask = (const int*)d_in[1];
  const float* Wq   = (const float*)d_in[2];
  const float* Wk   = (const float*)d_in[3];
  const float* Wv   = (const float*)d_in[4];
  const float* Wout = (const float*)d_in[5];
  float* out = (float*)d_out;

  // workspace (bf16): q 8MB | k 8MB | vT 8MB | O 8MB | x 8MB | 4 weights 2MB
  const size_t HEADELEMS = (size_t)HEADS * BATCH * NTOK * DHEAD;  // 4,194,304
  const size_t WELEMS = (size_t)DIM * DIM;                        // 262,144
  __bf16* qb  = (__bf16*)d_ws;
  __bf16* kbf = qb + HEADELEMS;
  __bf16* vtb = kbf + HEADELEMS;
  __bf16* ob  = vtb + HEADELEMS;
  __bf16* xb  = ob + HEADELEMS;
  __bf16* wqb = xb + (size_t)TOKS * DIM;
  __bf16* wkb = wqb + WELEMS;
  __bf16* wvb = wkb + WELEMS;
  __bf16* wob = wvb + WELEMS;

  // one-time bf16 conversion (x: 2048 blocks; each weight: 128 blocks)
  xattn_cvt_kernel<<<2048, 256, 0, stream>>>(x, xb, (int)((size_t)TOKS * DIM));
  xattn_cvt_kernel<<<128, 256, 0, stream>>>(Wq, wqb, (int)WELEMS);
  xattn_cvt_kernel<<<128, 256, 0, stream>>>(Wk, wkb, (int)WELEMS);
  xattn_cvt_kernel<<<128, 256, 0, stream>>>(Wv, wvb, (int)WELEMS);
  xattn_cvt_kernel<<<128, 256, 0, stream>>>(Wout, wob, (int)WELEMS);

  // 512 t-tiles * 24 head-weight groups = 12288 waves -> 1536 blocks of 8 waves
  xattn_qkv_kernel<<<1536, 256, 0, stream>>>(xb, wqb, wkb, wvb, qb, kbf, vtb);
  // 8 heads * 4 batch * 128 t-tiles = 4096 waves -> 512 blocks
  xattn_flash_kernel<<<512, 256, 0, stream>>>(qb, kbf, vtb, mask, ob);
  // 512 t-tiles * 8 e-groups = 4096 waves -> 512 blocks
  xattn_proj_kernel<<<512, 256, 0, stream>>>(ob, wob, out);
}
